// MultiHeadAttention_45346264711539
// MI455X (gfx1250) — matrix operand order repaired and checked by reading, not compiled, not run on hardware
//
#include <hip/hip_runtime.h>
#include <stdint.h>
#include <stddef.h>

// ---------------------------------------------------------------- types
typedef __attribute__((ext_vector_type(16))) __bf16 v16bf;
typedef __attribute__((ext_vector_type(8)))  float  v8f;
typedef int v4i __attribute__((vector_size(16)));   // GCC-style, matches builtin

#define DEVI __device__ __forceinline__

// problem constants
#define BATCH   4
#define SEQ     2048
#define DIN     1024
#define NH      16
#define HD      64
#define PROJ    1024
#define NQKV    3072
#define ROWS    (BATCH * SEQ)      // 8192
#define BH      (BATCH * NH)       // 64

// ---------------------------------------------------------------- async copy
#if defined(__has_builtin)
#if __has_builtin(__builtin_amdgcn_global_load_async_to_lds_b128)
#define HAVE_ASYNC_LDS 1
#endif
#endif

// dst = LDS (16B aligned), src = global (16B aligned), 16 bytes per lane
DEVI void cp_lds_b128(void* dst_lds, const void* src_global) {
#ifdef HAVE_ASYNC_LDS
  __builtin_amdgcn_global_load_async_to_lds_b128(
      (__attribute__((address_space(1))) v4i*)(src_global),
      (__attribute__((address_space(3))) v4i*)(dst_lds), 0, 0);
#else
  *(uint4*)dst_lds = *(const uint4*)src_global;
#endif
}

DEVI void cp_lds_wait() {
#ifdef HAVE_ASYNC_LDS
  asm volatile("s_wait_asynccnt 0x0" ::: "memory");
#endif
}

// ---------------------------------------------------------------- helpers
DEVI uint16_t f32_to_bf16(float f) {
  union { float f; uint32_t u; } cv; cv.f = f;
  const uint32_t x = cv.u;
  const uint32_t rne = (x + 0x7fffu + ((x >> 16) & 1u)) >> 16;
  const uint32_t nanv = (x >> 16) | 0x0040u;
  const bool is_nan = (x & 0x7fffffffu) > 0x7f800000u;
  return (uint16_t)(is_nan ? nanv : rne);     // branchless -> v_cndmask
}

DEVI v8f wmma_bf16(const uint32_t a[8], const uint32_t b[8], v8f c) {
  union { uint32_t u[8]; v16bf v; } A, B;
#pragma unroll
  for (int i = 0; i < 8; ++i) { A.u[i] = a[i]; B.u[i] = b[i]; }
  return __builtin_amdgcn_wmma_f32_16x16x32_bf16(false, A.v, false, B.v,
                                                 (short)0, c, false, false);
}

DEVI v8f vzero8() { v8f z = {0.f,0.f,0.f,0.f,0.f,0.f,0.f,0.f}; return z; }

// A fragment (16x32 bf16, row-major src, ld in elements)
// lane: m = lane&15 ; dword v: k = (v>>2)*16 + (lane>>4)*8 + (v&3)*2
DEVI void load_a_frag(const uint16_t* __restrict__ base, int ld, int m,
                      int k0, int half, uint32_t a[8]) {
  const uint16_t* p = base + (size_t)m * ld + k0;
#pragma unroll
  for (int v = 0; v < 8; ++v) {
    int kk = ((v >> 2) << 4) + (half << 3) + ((v & 3) << 1);
    a[v] = *(const uint32_t*)(p + kk);
  }
}

// ---------------------------------------------------------------- pack kernels
__global__ void pack_bf16_kernel(const float* __restrict__ src,
                                 uint16_t* __restrict__ dst, int n) {
  int i = blockIdx.x * blockDim.x + threadIdx.x;
  if (i < n) dst[i] = f32_to_bf16(src[i]);
}

__global__ void pack_qkv_kernel(const float* __restrict__ wq,
                                const float* __restrict__ wk,
                                const float* __restrict__ wv,
                                const float* __restrict__ bq,
                                const float* __restrict__ bk,
                                const float* __restrict__ bv,
                                uint16_t* __restrict__ wqkv,   // [DIN][NQKV]
                                float* __restrict__ bqkv) {    // [NQKV]
  int i = blockIdx.x * blockDim.x + threadIdx.x;
  if (i < DIN * NQKV) {
    int row = i / NQKV;
    int col = i - row * NQKV;
    int w = col >> 10;          // 0=Q 1=K 2=V
    int c = col & 1023;
    const float* s = (w == 0) ? wq : ((w == 1) ? wk : wv);
    wqkv[i] = f32_to_bf16(s[row * PROJ + c]);
  }
  if (i < NQKV) {
    int w = i >> 10, c = i & 1023;
    const float* sb = (w == 0) ? bq : ((w == 1) ? bk : bv);
    bqkv[i] = sb[c];
  }
}

// ---------------------------------------------------------------- QKV GEMM
// M=8192, N=3072, K=1024. One wave -> 16x64 tile, register double-buffered
// over k-steps so loads for step i+1 overlap the WMMAs of step i.
__global__ void __launch_bounds__(256)
qkv_gemm_kernel(const uint16_t* __restrict__ xb,     // [ROWS][DIN]
                const uint16_t* __restrict__ wqkv,   // [DIN][NQKV]
                const float*    __restrict__ bqkv,   // [NQKV]
                uint16_t* __restrict__ qb,           // [BH][SEQ][HD]
                uint16_t* __restrict__ ktb,          // [BH][HD][SEQ]
                uint16_t* __restrict__ vb) {         // [BH][SEQ][HD]
  const int lane = threadIdx.x & 31;
  const int wave = threadIdx.x >> 5;
  const int flat = blockIdx.x * 8 + wave;            // 24576 wave tiles
  const int ntile = flat % (NQKV / 64);              // 0..47
  const int mtile = flat / (NQKV / 64);              // 0..511
  const int half = lane >> 4;
  const int lm = lane & 15;

  v8f acc[4] = {vzero8(), vzero8(), vzero8(), vzero8()};

  const uint16_t* arow = xb + (size_t)(mtile * 16) * DIN;

  uint32_t a[2][8];
  uint32_t bfr[2][4][8];

  auto load_step = [&](int slot, int k0) {
    load_a_frag(arow, DIN, lm, k0, half, a[slot]);
    const uint16_t* bbase = wqkv + (size_t)k0 * NQKV + ntile * 64 + lm;
    __builtin_prefetch(bbase + 64 * NQKV, 0, 1);
#pragma unroll
    for (int nf = 0; nf < 4; ++nf)
#pragma unroll
      for (int v = 0; v < 8; ++v) {
        const int kk = ((v >> 2) << 4) + (half << 3) + ((v & 3) << 1);
        bfr[slot][nf][v] = (uint32_t)bbase[(size_t)kk * NQKV + nf * 16] |
                           ((uint32_t)bbase[(size_t)(kk + 1) * NQKV + nf * 16] << 16);
      }
  };

  load_step(0, 0);
  int slot = 0;
  for (int k0 = 0; k0 < DIN; k0 += 32) {
    if (k0 + 32 < DIN) load_step(slot ^ 1, k0 + 32);
#pragma unroll
    for (int nf = 0; nf < 4; ++nf)
      acc[nf] = wmma_bf16(a[slot], bfr[slot][nf], acc[nf]);
    slot ^= 1;
  }

  // epilogue: bias + scatter. which weight is uniform per wave (64 | 1024).
  const int which = ntile >> 4;                      // 0=Q 1=K 2=V
#pragma unroll
  for (int nf = 0; nf < 4; ++nf) {
    const int n = ntile * 64 + nf * 16 + lm;         // 0..3071
    const int c = n & 1023;
    const int h = c >> 6;
    const int d = c & 63;
    const float bias = bqkv[n];
#pragma unroll
    for (int v = 0; v < 8; ++v) {
      const int srow = mtile * 16 + v + 8 * half;    // 0..8191
      const int b_ = srow >> 11;                     // batch
      const int s_ = srow & 2047;                    // seq pos
      const int bh = b_ * NH + h;
      const uint16_t val = f32_to_bf16(acc[nf][v] + bias);
      if (which == 0)
        qb[((size_t)bh * SEQ + s_) * HD + d] = val;
      else if (which == 1)
        ktb[((size_t)bh * HD + d) * SEQ + s_] = val;
      else
        vb[((size_t)bh * SEQ + s_) * HD + d] = val;
    }
  }
}

// ---------------------------------------------------------------- attention
// Block = 128 threads (4 waves) sharing one (b,h); each wave owns 16 query
// rows. K/V tiles (64 keys) staged in LDS (double-buffered) via async
// global->LDS; all 4 waves consume them with ds loads. Flash-style softmax.
__global__ void __launch_bounds__(128)
attn_kernel(const uint16_t* __restrict__ qb,   // [BH][SEQ][HD]
            const uint16_t* __restrict__ ktb,  // [BH][HD][SEQ]
            const uint16_t* __restrict__ vb,   // [BH][SEQ][HD]
            uint16_t* __restrict__ ctx) {      // [B][S][H*D] bf16
  __shared__ __align__(16) uint16_t kbuf[2][64 * 64];  // [d][key] 8KB x2
  __shared__ __align__(16) uint16_t vbuf[2][64 * 64];  // [key][d] 8KB x2
  __shared__ __align__(16) uint16_t pbuf[4][16 * 64];  // per-wave P tile

  const int tid  = threadIdx.x;
  const int lane = tid & 31;
  const int wave = tid >> 5;
  const int half = lane >> 4;
  const int lm = lane & 15;

  const int bh = blockIdx.x >> 5;                     // 0..63
  const int qt = ((blockIdx.x & 31) << 2) + wave;     // 0..127 (16-row tiles)

  const uint16_t* qbase  = qb  + (size_t)bh * SEQ * HD;
  const uint16_t* ktbase = ktb + (size_t)bh * HD * SEQ;
  const uint16_t* vbase  = vb  + (size_t)bh * SEQ * HD;

  // stage one 64-key tile: 512 x 16B chunks for K and for V, 128 threads
  auto stage_tile = [&](int buf, int kt0) {
#pragma unroll
    for (int i = 0; i < 4; ++i) {
      const int idx = i * 128 + tid;    // 0..511
      const int row = idx >> 3;         // 0..63
      const int c8  = (idx & 7) << 3;   // element offset in row (x8 elems)
      cp_lds_b128(&kbuf[buf][row * 64 + c8], ktbase + (size_t)row * SEQ + kt0 + c8);
      cp_lds_b128(&vbuf[buf][row * 64 + c8], vbase + (size_t)(kt0 + row) * HD + c8);
    }
  };

  // Q A-fragments resident for the whole loop (16 x 64 = two k-steps)
  uint32_t qa[2][8];
#pragma unroll
  for (int ks = 0; ks < 2; ++ks)
    load_a_frag(qbase + (size_t)(qt * 16) * HD, HD, lm, ks * 32, half, qa[ks]);

  v8f cacc[4] = {vzero8(), vzero8(), vzero8(), vzero8()};
  float rowmax[8], rowsum[8];
#pragma unroll
  for (int v = 0; v < 8; ++v) { rowmax[v] = -1e30f; rowsum[v] = 0.f; }

  const float scale = 0.125f;                         // 1/sqrt(64)
  uint32_t* p32 = (uint32_t*)&pbuf[wave][0];

  stage_tile(0, 0);
  cp_lds_wait();
  __syncthreads();

  int cur = 0;
  for (int kt0 = 0; kt0 < SEQ; kt0 += 64) {
    if (kt0 + 64 < SEQ) stage_tile(cur ^ 1, kt0 + 64);  // overlap with compute

    const uint32_t* k32 = (const uint32_t*)&kbuf[cur][0];
    const uint32_t* v32 = (const uint32_t*)&vbuf[cur][0];

    // ---- scores: S = Q (16x64) x K^T (64x64), B-frags from LDS
    v8f sfr[4] = {vzero8(), vzero8(), vzero8(), vzero8()};
#pragma unroll
    for (int ks = 0; ks < 2; ++ks) {
#pragma unroll
      for (int nf = 0; nf < 4; ++nf) {
        uint32_t b[8];
#pragma unroll
        for (int v = 0; v < 8; ++v) {
          const int d = ks * 32 + ((v >> 2) << 4) + (half << 3) + ((v & 3) << 1);
          b[v] = (uint32_t)kbuf[cur][d * 64 + nf * 16 + lm] |
                 ((uint32_t)kbuf[cur][(d + 1) * 64 + nf * 16 + lm] << 16);
        }
        sfr[nf] = wmma_bf16(qa[ks], b, sfr[nf]);
      }
    }

    // ---- online softmax (rows live across the 16 lanes of this half)
#pragma unroll
    for (int nf = 0; nf < 4; ++nf)
#pragma unroll
      for (int v = 0; v < 8; ++v) sfr[nf][v] *= scale;

#pragma unroll
    for (int v = 0; v < 8; ++v) {
      float mloc = fmaxf(fmaxf(sfr[0][v], sfr[1][v]),
                         fmaxf(sfr[2][v], sfr[3][v]));
#pragma unroll
      for (int off = 1; off < 16; off <<= 1)
        mloc = fmaxf(mloc, __shfl_xor(mloc, off, 32));
      const float nm = fmaxf(rowmax[v], mloc);
      const float sc = __expf(rowmax[v] - nm);
      rowmax[v] = nm;
      float psum = 0.f;
#pragma unroll
      for (int nf = 0; nf < 4; ++nf) {
        const float p = __expf(sfr[nf][v] - nm);
        sfr[nf][v] = p;
        psum += p;
      }
#pragma unroll
      for (int off = 1; off < 16; off <<= 1)
        psum += __shfl_xor(psum, off, 32);
      rowsum[v] = rowsum[v] * sc + psum;
#pragma unroll
      for (int nf = 0; nf < 4; ++nf) cacc[nf][v] *= sc;
    }

    // ---- P (D-layout) -> LDS -> A-layout
#pragma unroll
    for (int nf = 0; nf < 4; ++nf)
#pragma unroll
      for (int v = 0; v < 8; ++v) {
        const int m = v + 8 * half;
        pbuf[wave][m * 64 + nf * 16 + lm] = f32_to_bf16(sfr[nf][v]);
      }
    __syncthreads();

    uint32_t pa[2][8];
#pragma unroll
    for (int ks = 0; ks < 2; ++ks)
#pragma unroll
      for (int v = 0; v < 8; ++v) {
        const int kk = ks * 32 + ((v >> 2) << 4) + (half << 3) + ((v & 3) << 1);
        pa[ks][v] = p32[(lm * 64 + kk) >> 1];
      }

    // ---- context: C += P (16x64) x V (64x64), B-frags from LDS
#pragma unroll
    for (int ks = 0; ks < 2; ++ks) {
#pragma unroll
      for (int nf = 0; nf < 4; ++nf) {
        uint32_t b[8];
#pragma unroll
        for (int v = 0; v < 8; ++v) {
          const int krow = ks * 32 + ((v >> 2) << 4) + (half << 3) + ((v & 3) << 1);
          b[v] = (uint32_t)vbuf[cur][krow * 64 + nf * 16 + lm] |
                 ((uint32_t)vbuf[cur][(krow + 1) * 64 + nf * 16 + lm] << 16);
        }
        cacc[nf] = wmma_bf16(pa[ks], b, cacc[nf]);
      }
    }

    cp_lds_wait();       // next tile fully landed in LDS
    __syncthreads();     // all waves done reading cur (and pbuf)
    cur ^= 1;
  }

  // ---- normalize + store ctx as [B][S][H*D] (bf16)
  const int b_ = bh >> 4;
  const int h_ = bh & 15;
#pragma unroll
  for (int nf = 0; nf < 4; ++nf) {
    const int d = nf * 16 + lm;
#pragma unroll
    for (int v = 0; v < 8; ++v) {
      const int m = v + 8 * half;
      const int s_ = qt * 16 + m;
      const float val = cacc[nf][v] / rowsum[v];
      ctx[((size_t)(b_ * SEQ + s_)) * PROJ + h_ * HD + d] = f32_to_bf16(val);
    }
  }
}

// ---------------------------------------------------------------- final proj
// out = ctx [8192x1024] @ Wf [1024x64] + bf, f32 out. One wave -> 16x64 tile,
// register double-buffered like the QKV GEMM.
__global__ void __launch_bounds__(256)
final_gemm_kernel(const uint16_t* __restrict__ ctx,  // [ROWS][PROJ]
                  const uint16_t* __restrict__ wfb,  // [PROJ][HD]
                  const float*    __restrict__ bf_,  // [HD]
                  float* __restrict__ out) {         // [ROWS][HD]
  const int lane = threadIdx.x & 31;
  const int wave = threadIdx.x >> 5;
  const int mtile = blockIdx.x * 8 + wave;           // 0..511
  const int half = lane >> 4;
  const int lm = lane & 15;

  v8f acc[4] = {vzero8(), vzero8(), vzero8(), vzero8()};
  const uint16_t* arow = ctx + (size_t)(mtile * 16) * PROJ;

  uint32_t a[2][8];
  uint32_t bfr[2][4][8];

  auto load_step = [&](int slot, int k0) {
    load_a_frag(arow, PROJ, lm, k0, half, a[slot]);
    const uint16_t* bbase = wfb + (size_t)k0 * HD + lm;
#pragma unroll
    for (int nf = 0; nf < 4; ++nf)
#pragma unroll
      for (int v = 0; v < 8; ++v) {
        const int kk = ((v >> 2) << 4) + (half << 3) + ((v & 3) << 1);
        bfr[slot][nf][v] = (uint32_t)bbase[kk * HD + nf * 16] |
                           ((uint32_t)bbase[(kk + 1) * HD + nf * 16] << 16);
      }
  };

  load_step(0, 0);
  int slot = 0;
  for (int k0 = 0; k0 < PROJ; k0 += 32) {
    if (k0 + 32 < PROJ) load_step(slot ^ 1, k0 + 32);
#pragma unroll
    for (int nf = 0; nf < 4; ++nf)
      acc[nf] = wmma_bf16(a[slot], bfr[slot][nf], acc[nf]);
    slot ^= 1;
  }

#pragma unroll
  for (int nf = 0; nf < 4; ++nf) {
    const int n = nf * 16 + lm;
    const float bias = bf_[n];
#pragma unroll
    for (int v = 0; v < 8; ++v) {
      const int srow = mtile * 16 + v + 8 * half;
      out[(size_t)srow * HD + n] = acc[nf][v] + bias;
    }
  }
}

// ---------------------------------------------------------------- launch
extern "C" void kernel_launch(void* const* d_in, const int* in_sizes, int n_in,
                              void* d_out, int out_size, void* d_ws, size_t ws_size,
                              hipStream_t stream) {
  (void)in_sizes; (void)n_in; (void)out_size; (void)ws_size;
  const float* x  = (const float*)d_in[0];
  const float* Wq = (const float*)d_in[1];
  const float* bq = (const float*)d_in[2];
  const float* Wk = (const float*)d_in[3];
  const float* bk = (const float*)d_in[4];
  const float* Wv = (const float*)d_in[5];
  const float* bv = (const float*)d_in[6];
  const float* Wf = (const float*)d_in[7];
  const float* bf = (const float*)d_in[8];
  float* out = (float*)d_out;

  char* w = (char*)d_ws;
  uint16_t* xb   = (uint16_t*)w; w += (size_t)ROWS * DIN * 2;     // 16 MB
  uint16_t* wqkv = (uint16_t*)w; w += (size_t)DIN * NQKV * 2;     // 6 MB
  float*    bqkv = (float*)w;    w += (size_t)NQKV * 4;           // 12 KB
  uint16_t* qb   = (uint16_t*)w; w += (size_t)BH * SEQ * HD * 2;  // 16 MB
  uint16_t* ktb  = (uint16_t*)w; w += (size_t)BH * HD * SEQ * 2;  // 16 MB
  uint16_t* vb   = (uint16_t*)w; w += (size_t)BH * SEQ * HD * 2;  // 16 MB
  uint16_t* ctx  = (uint16_t*)w; w += (size_t)ROWS * PROJ * 2;    // 16 MB
  uint16_t* wfb  = (uint16_t*)w; w += (size_t)PROJ * HD * 2;      // 128 KB

  // 1) pack activations & weights to bf16
  {
    int n = ROWS * DIN;
    pack_bf16_kernel<<<(n + 255) / 256, 256, 0, stream>>>(x, xb, n);
  }
  pack_qkv_kernel<<<(DIN * NQKV + 255) / 256, 256, 0, stream>>>(
      Wq, Wk, Wv, bq, bk, bv, wqkv, bqkv);
  {
    int n = PROJ * HD;
    pack_bf16_kernel<<<(n + 255) / 256, 256, 0, stream>>>(Wf, wfb, n);
  }

  // 2) fused QKV projection (K stored transposed)
  qkv_gemm_kernel<<<3072, 256, 0, stream>>>(xb, wqkv, bqkv, qb, ktb, vb);

  // 3) flash attention (LDS-staged K/V, async copies, 4 waves share tiles)
  attn_kernel<<<2048, 128, 0, stream>>>(qb, ktb, vb, ctx);

  // 4) output projection
  final_gemm_kernel<<<64, 256, 0, stream>>>(ctx, wfb, bf, out);
}
